// GraphEncoder_57801669869819
// MI455X (gfx1250) — matrix operand order repaired and checked by reading, not compiled, not run on hardware
//
#include <hip/hip_runtime.h>
#include <hip/hip_bf16.h>

// ---------------------------------------------------------------------------
// GraphEncoder for gfx1250 (MI455X): all GEMM-shaped work via v_wmma_f32_16x16x32_f16
// ---------------------------------------------------------------------------

typedef __attribute__((ext_vector_type(16))) _Float16 v16h;
typedef __attribute__((ext_vector_type(8)))  _Float16 v8h;
typedef __attribute__((ext_vector_type(8)))  float    v8f;

#define BN_TOT  4096      // B*N nodes
#define EMB     128
#define NHEAD   8
#define DHEAD   16
#define NEDGE   16384
#define NGRAPH  32
#define NNODE   128
#define NLAYER  4
#define EIN2    48        // 15 bond + 32 rbf + 1 bias column
#define KEDGE   (EIN2*128)  // 6144: K-dim of the edge-network GEMM

static __device__ __forceinline__ v8f wmma16(v16h a, v16h b, v8f c) {
  // D = A(16x32 f16) x B(32x16 f16) + C(16x16 f32)
  return __builtin_amdgcn_wmma_f32_16x16x32_f16(false, a, false, b, (short)0, c,
                                                false, false);
}

static __device__ __forceinline__ v8f vzero8() {
  v8f z;
  #pragma unroll
  for (int i = 0; i < 8; ++i) z[i] = 0.f;
  return z;
}

// ---------------------------------------------------------------------------
// Weight prep: W[K,N] f32 -> Wt[N,K] f16 (B stored transposed for coalesced tiles)
// ---------------------------------------------------------------------------
__global__ void transpose_conv_kernel(const float* __restrict__ W,
                                      _Float16* __restrict__ Wt, int K, int N) {
  int n  = blockIdx.x * 16 + threadIdx.x;
  int kk = blockIdx.y * 16 + threadIdx.y;
  if (n < N && kk < K) Wt[(size_t)n * K + kk] = (_Float16)W[(size_t)kk * N + n];
}

// Edge-network B matrix, transposed: BmatT[e][c*128+f] = Wlin[c][e*128+f] (c<47), blin[e*128+f] (c==47)
__global__ void build_edgeB_kernel(const float* __restrict__ Wlin,
                                   const float* __restrict__ blin,
                                   _Float16* __restrict__ BmatT) {
  int idx = blockIdx.x * 256 + threadIdx.x;      // 128 * 6144 total
  int e   = idx / KEDGE;
  int rem = idx - e * KEDGE;
  int c   = rem >> 7;
  int f   = rem & 127;
  float w = (c < 47) ? Wlin[(size_t)c * 16384 + e * 128 + f] : blin[e * 128 + f];
  BmatT[idx] = (_Float16)w;
}

// Gaussian RBF expansion + bias column: ea[M,16] -> ea2[M,48]
__global__ void rbf_kernel(const float* __restrict__ ea, float* __restrict__ ea2) {
  int m = blockIdx.x * 256 + threadIdx.x;
  if (m >= NEDGE) return;
  const float* e = ea + (size_t)m * 16;
  float* o = ea2 + (size_t)m * EIN2;
  #pragma unroll
  for (int j = 0; j < 15; ++j) o[j] = e[j];
  float d = fminf(10.f, fmaxf(0.f, e[15]));
  const float gamma = (31.f / 5.f) * (31.f / 5.f);   // (CUTOFF/(K-1))^-2
  #pragma unroll
  for (int kk = 0; kk < 32; ++kk) {
    float c = 5.f * (float)kk / 31.f;
    float t = d - c;
    o[15 + kk] = __expf(-gamma * t * t);
  }
  o[47] = 1.f;                                        // bias (blin) column
}

__global__ void zero_kernel(float* __restrict__ p, int n) {
  int i = blockIdx.x * 256 + threadIdx.x;
  if (i < n) p[i] = 0.f;
}

// Pairwise distances db[b][i][j]
__global__ void pairdist_kernel(const float* __restrict__ pos, float* __restrict__ db) {
  int idx = blockIdx.x * 256 + threadIdx.x;        // B*N*N = 524288
  int b = idx >> 14, rem = idx & 16383;
  int i = rem >> 7, j = rem & 127;
  const float* pi = pos + (size_t)(b * NNODE + i) * 3;
  const float* pj = pos + (size_t)(b * NNODE + j) * 3;
  float dx = pi[0] - pj[0], dy = pi[1] - pj[1], dz = pi[2] - pj[2];
  float d2 = dx * dx + dy * dy + dz * dz;
  db[idx] = d2 > 0.f ? sqrtf(d2) : 0.f;
}

// out = LN(x + y) * g + b   (rows of 128; one wave32 per row, 4 elems/lane)
__global__ void add_ln_kernel(const float* __restrict__ x, const float* __restrict__ y,
                              const float* __restrict__ g, const float* __restrict__ bb,
                              float* __restrict__ out) {
  int lane = threadIdx.x & 31, wave = threadIdx.x >> 5;
  int row  = blockIdx.x * 8 + wave;
  size_t base = (size_t)row * EMB;
  float v[4]; float s = 0.f;
  #pragma unroll
  for (int j = 0; j < 4; ++j) { v[j] = x[base + lane + 32 * j] + y[base + lane + 32 * j]; s += v[j]; }
  #pragma unroll
  for (int off = 16; off; off >>= 1) s += __shfl_xor(s, off);
  float mean = s * (1.f / 128.f);
  float q = 0.f;
  #pragma unroll
  for (int j = 0; j < 4; ++j) { float d = v[j] - mean; q += d * d; }
  #pragma unroll
  for (int off = 16; off; off >>= 1) q += __shfl_xor(q, off);
  float rstd = rsqrtf(q * (1.f / 128.f) + 1e-5f);
  #pragma unroll
  for (int j = 0; j < 4; ++j) {
    int e = lane + 32 * j;
    out[base + e] = (v[j] - mean) * rstd * g[e] + bb[e];
  }
}

// Mean-pool per graph: out[b][e] = sum_i xp[b*N+i][e] / (N + 1e-8)
__global__ void pool_kernel(const float* __restrict__ xp, float* __restrict__ out) {
  int b = blockIdx.x, e = threadIdx.x;
  float s = 0.f;
  for (int i = 0; i < NNODE; ++i) s += xp[(size_t)(b * NNODE + i) * EMB + e];
  out[b * EMB + e] = s / (128.f + 1e-8f);
}

// ---------------------------------------------------------------------------
// Generic WMMA GEMM: Out[M,N] = act(A[M,K] f32 @ Bt[N,K]^T f16 + bias[N])
// 128 threads (4 waves); block tile 64x64; each wave: 16 rows x 64 cols.
// ---------------------------------------------------------------------------
template <bool RELU>
__global__ __launch_bounds__(128)
void gemm_kernel(const float* __restrict__ A, const _Float16* __restrict__ Bt,
                 const float* __restrict__ bias, float* __restrict__ Out,
                 int Kdim, int Ndim) {
  __shared__ __align__(16) _Float16 As[64][40];
  __shared__ __align__(16) _Float16 Bs[64][40];
  const int tid = threadIdx.x, lane = tid & 31, wave = tid >> 5;
  const int r = lane & 15, hi = lane >> 4;
  const int rowBase = blockIdx.y * 64, nBase = blockIdx.x * 64;
  v8f acc[4];
  #pragma unroll
  for (int i = 0; i < 4; ++i) acc[i] = vzero8();
  const int ar = tid >> 1, ac = (tid & 1) * 16;
  const int nsteps = Kdim >> 5;
  for (int ks = 0; ks < nsteps; ++ks) {
    __syncthreads();
    { // A tile 64x32 (f32 -> f16)
      const float* ap = A + (size_t)(rowBase + ar) * Kdim + ks * 32 + ac;
      float4 f0 = *(const float4*)(ap + 0), f1 = *(const float4*)(ap + 4);
      float4 f2 = *(const float4*)(ap + 8), f3 = *(const float4*)(ap + 12);
      _Float16* as = &As[ar][ac];
      as[0] = (_Float16)f0.x; as[1] = (_Float16)f0.y; as[2] = (_Float16)f0.z; as[3] = (_Float16)f0.w;
      as[4] = (_Float16)f1.x; as[5] = (_Float16)f1.y; as[6] = (_Float16)f1.z; as[7] = (_Float16)f1.w;
      as[8] = (_Float16)f2.x; as[9] = (_Float16)f2.y; as[10] = (_Float16)f2.z; as[11] = (_Float16)f2.w;
      as[12] = (_Float16)f3.x; as[13] = (_Float16)f3.y; as[14] = (_Float16)f3.z; as[15] = (_Float16)f3.w;
    }
    { // B tile (pre-transposed, f16): 64 n-rows x 32 k
      const _Float16* bp = Bt + (size_t)(nBase + ar) * Kdim + ks * 32 + ac;
      *(uint4*)&Bs[ar][ac]     = *(const uint4*)bp;
      *(uint4*)&Bs[ar][ac + 8] = *(const uint4*)(bp + 8);
    }
    __syncthreads();
    union { v16h v; v8h h[2]; } au;
    au.h[0] = *(const v8h*)&As[wave * 16 + r][hi * 8];
    au.h[1] = *(const v8h*)&As[wave * 16 + r][hi * 8 + 16];
    #pragma unroll
    for (int nb = 0; nb < 4; ++nb) {
      union { v16h v; v8h h[2]; } bu;
      bu.h[0] = *(const v8h*)&Bs[nb * 16 + r][hi * 8];
      bu.h[1] = *(const v8h*)&Bs[nb * 16 + r][hi * 8 + 16];
      acc[nb] = wmma16(au.v, bu.v, acc[nb]);
    }
  }
  #pragma unroll
  for (int nb = 0; nb < 4; ++nb) {
    int col = nBase + nb * 16 + r;
    float bb = bias[col];
    #pragma unroll
    for (int reg = 0; reg < 8; ++reg) {
      int row = rowBase + wave * 16 + reg + hi * 8;
      float v = acc[nb][reg] + bb;
      if (RELU) v = v > 0.f ? v : 0.f;
      Out[(size_t)row * Ndim + col] = v;
    }
  }
}

// ---------------------------------------------------------------------------
// Edge-message GEMM: msg[m,e] = sum_{c,f} ea2[m,c] * h[ei1[m],f] * BmatT[e][c*128+f]
// A fragments synthesized on the fly; result scattered to agg[ei0[m]] with atomics.
// Block: 64 edges x 128 output cols, 192 K-steps of 32.
// ---------------------------------------------------------------------------
__global__ __launch_bounds__(128)
void edge_msg_kernel(const float* __restrict__ ea2, const float* __restrict__ h,
                     const int* __restrict__ ei, const _Float16* __restrict__ BmatT,
                     float* __restrict__ agg) {
  __shared__ __align__(16) _Float16 Hs[64][136];   // gathered h_j rows, f16
  __shared__ __align__(16) float    Es[64][48];    // edge feature rows
  __shared__ __align__(16) _Float16 Bs[128][40];   // per-step B tile: 128 n x 32 k
  __shared__ int dstN[64];
  const int tid = threadIdx.x, lane = tid & 31, wave = tid >> 5;
  const int r = lane & 15, hi = lane >> 4;
  const int ebase = blockIdx.x * 64;
  const int* ei0 = ei;
  const int* ei1 = ei + NEDGE;
  if (tid < 64) dstN[tid] = ei0[ebase + tid];
  {
    int row = tid >> 1, cb = (tid & 1) * 64;
    int src = ei1[ebase + row];
    const float* hp = h + (size_t)src * EMB + cb;
    #pragma unroll
    for (int j = 0; j < 64; j += 4) {
      float4 f = *(const float4*)(hp + j);
      Hs[row][cb + j + 0] = (_Float16)f.x; Hs[row][cb + j + 1] = (_Float16)f.y;
      Hs[row][cb + j + 2] = (_Float16)f.z; Hs[row][cb + j + 3] = (_Float16)f.w;
    }
    int ccb = (tid & 1) * 24;
    const float* ep = ea2 + (size_t)(ebase + row) * EIN2 + ccb;
    #pragma unroll
    for (int j = 0; j < 24; ++j) Es[row][ccb + j] = ep[j];
  }
  v8f acc[8];
  #pragma unroll
  for (int i = 0; i < 8; ++i) acc[i] = vzero8();
  const int arow = wave * 16 + r;
  for (int step = 0; step < 192; ++step) {
    __syncthreads();
    { // stage B tile: row n = tid, 32 halves from BmatT[n][step*32 ..]
      const _Float16* bp = BmatT + (size_t)tid * KEDGE + step * 32;
      uint4* d = (uint4*)&Bs[tid][0];
      const uint4* s = (const uint4*)bp;
      d[0] = s[0]; d[1] = s[1]; d[2] = s[2]; d[3] = s[3];
    }
    __syncthreads();
    int c  = step >> 2;
    int fb = (step & 3) << 5;
    _Float16 sc = (_Float16)Es[arow][c];
    union { v16h v; v8h h[2]; } au;
    au.h[0] = *(const v8h*)&Hs[arow][fb + hi * 8];
    au.h[1] = *(const v8h*)&Hs[arow][fb + hi * 8 + 16];
    v16h a;
    #pragma unroll
    for (int j = 0; j < 16; ++j) a[j] = au.v[j] * sc;   // ea2[m,c] * h_j[m,f]
    #pragma unroll
    for (int nb = 0; nb < 8; ++nb) {
      union { v16h v; v8h h[2]; } bu;
      bu.h[0] = *(const v8h*)&Bs[nb * 16 + r][hi * 8];
      bu.h[1] = *(const v8h*)&Bs[nb * 16 + r][hi * 8 + 16];
      acc[nb] = wmma16(a, bu.v, acc[nb]);
    }
  }
  #pragma unroll
  for (int nb = 0; nb < 8; ++nb) {
    int col = nb * 16 + r;
    #pragma unroll
    for (int reg = 0; reg < 8; ++reg) {
      int rl = wave * 16 + reg + hi * 8;
      atomicAdd(&agg[(size_t)dstN[rl] * EMB + col], acc[nb][reg]);
    }
  }
}

// ---------------------------------------------------------------------------
// Attention: one block per (graph b, head hh). S = Q K^T / sqrt(dh) + db, clip,
// softmax, O = P V.  K=16 padded to 32 with zeroed upper-A halves.
// Scores staged in global scratch (sc) to keep static LDS < 64KB.
// ---------------------------------------------------------------------------
__global__ __launch_bounds__(128)
void attn_kernel(const float* __restrict__ q, const float* __restrict__ k,
                 const float* __restrict__ v, const float* __restrict__ db,
                 float* __restrict__ sc, float* __restrict__ o) {
  __shared__ __align__(16) _Float16 Qs[128][24];
  __shared__ __align__(16) _Float16 Ks[128][32];
  __shared__ __align__(16) _Float16 Vt[16][136];   // transposed V: [d][j]
  __shared__ __align__(16) _Float16 P[128][136];   // softmax probs (f16 A matrix)
  const int b = blockIdx.x >> 3, hh = blockIdx.x & 7;
  const int tid = threadIdx.x, lane = tid & 31, wave = tid >> 5;
  const int r = lane & 15, hi = lane >> 4;
  const float inv = 0.25f;   // 1/sqrt(DH)
  {
    int i = tid;
    const float* qp = q + (size_t)(b * NNODE + i) * EMB + hh * DHEAD;
    const float* kp = k + (size_t)(b * NNODE + i) * EMB + hh * DHEAD;
    const float* vp = v + (size_t)(b * NNODE + i) * EMB + hh * DHEAD;
    #pragma unroll
    for (int d = 0; d < 16; ++d) {
      Qs[i][d] = (_Float16)qp[d];
      Ks[i][d] = (_Float16)kp[d];
      Vt[d][i] = (_Float16)vp[d];
    }
    #pragma unroll
    for (int d = 16; d < 24; ++d) { Qs[i][d] = (_Float16)0.f; }
  }
  __syncthreads();
  float* scb = sc + (size_t)blockIdx.x * NNODE * NNODE;
  // --- S = Q K^T: 8x8 grid of 16x16 tiles, 16 tiles per wave ---
  for (int t = 0; t < 16; ++t) {
    int idx = wave * 16 + t;
    int ti = idx >> 3, tj = idx & 7;
    union { v16h v; v8h h[2]; } au, bu;
    au.h[0] = *(const v8h*)&Qs[ti * 16 + r][hi * 8];       // K 0..15 (real)
    #pragma unroll
    for (int j = 0; j < 8; ++j) au.h[1][j] = (_Float16)0.f; // K 16..31 zero-pad
    bu.h[0] = *(const v8h*)&Ks[tj * 16 + r][hi * 8];       // K 0..15 (real)
    bu.h[1] = *(const v8h*)&Ks[tj * 16 + r][hi * 8 + 16];  // K>=16 garbage * 0 = 0
    v8f c = wmma16(au.v, bu.v, vzero8());
    #pragma unroll
    for (int reg = 0; reg < 8; ++reg) {
      int i = ti * 16 + reg + hi * 8;
      int j = tj * 16 + r;
      float val = c[reg] * inv + db[((size_t)b * NNODE + i) * NNODE + j];
      val = fminf(10.f, fmaxf(-10.f, val));
      scb[(size_t)i * NNODE + j] = val;
    }
  }
  __syncthreads();
  { // --- softmax per row (one thread per row) ---
    int i = tid;
    const float* srow = scb + (size_t)i * NNODE;
    float mx = -1e30f;
    for (int j = 0; j < NNODE; ++j) mx = fmaxf(mx, srow[j]);
    float s = 0.f;
    for (int j = 0; j < NNODE; ++j) {
      float e = __expf(srow[j] - mx);
      s += e;
      P[i][j] = (_Float16)e;
    }
    _Float16 is = (_Float16)(1.f / s);
    for (int j = 0; j < NNODE; ++j) P[i][j] = P[i][j] * is;
  }
  __syncthreads();
  // --- O = P V: 8 row tiles x 1 col tile (DH=16), K=128 in 4 steps ---
  for (int t = 0; t < 2; ++t) {
    int ti = wave * 2 + t;
    v8f c = vzero8();
    #pragma unroll
    for (int ks = 0; ks < 4; ++ks) {
      union { v16h v; v8h h[2]; } au, bu;
      au.h[0] = *(const v8h*)&P[ti * 16 + r][ks * 32 + hi * 8];
      au.h[1] = *(const v8h*)&P[ti * 16 + r][ks * 32 + hi * 8 + 16];
      bu.h[0] = *(const v8h*)&Vt[r][ks * 32 + hi * 8];
      bu.h[1] = *(const v8h*)&Vt[r][ks * 32 + hi * 8 + 16];
      c = wmma16(au.v, bu.v, c);
    }
    #pragma unroll
    for (int reg = 0; reg < 8; ++reg) {
      int i = ti * 16 + reg + hi * 8;
      o[(size_t)(b * NNODE + i) * EMB + hh * DHEAD + r] = c[reg];
    }
  }
}

// ---------------------------------------------------------------------------
// Host orchestration
// ---------------------------------------------------------------------------
extern "C" void kernel_launch(void* const* d_in, const int* in_sizes, int n_in,
                              void* d_out, int out_size, void* d_ws, size_t ws_size,
                              hipStream_t stream) {
  (void)in_sizes; (void)n_in; (void)out_size; (void)ws_size;
  const float* x     = (const float*)d_in[0];
  const int*   ei    = (const int*)  d_in[1];
  const float* ea    = (const float*)d_in[2];
  const float* pos   = (const float*)d_in[3];
  /* d_in[4] = batch (contiguous equal graphs -> unused) */
  const float* Wproj = (const float*)d_in[5];
  const float* bproj = (const float*)d_in[6];
  const float* Wlin  = (const float*)d_in[7];
  const float* blin  = (const float*)d_in[8];
  const float* g_en  = (const float*)d_in[9];
  const float* b_en  = (const float*)d_in[10];
  const float* Wq    = (const float*)d_in[11];
  const float* Wk    = (const float*)d_in[12];
  const float* Wv    = (const float*)d_in[13];
  const float* Wo    = (const float*)d_in[14];
  const float* bq    = (const float*)d_in[15];
  const float* bk    = (const float*)d_in[16];
  const float* bv    = (const float*)d_in[17];
  const float* bo    = (const float*)d_in[18];
  const float* g1    = (const float*)d_in[19];
  const float* b1    = (const float*)d_in[20];
  const float* W1    = (const float*)d_in[21];
  const float* bf1   = (const float*)d_in[22];
  const float* W2    = (const float*)d_in[23];
  const float* bf2   = (const float*)d_in[24];
  const float* g2    = (const float*)d_in[25];
  const float* b2    = (const float*)d_in[26];

  char* ws = (char*)d_ws;
  size_t off = 0;
  auto alloc = [&](size_t bytes) {
    size_t o = off; off += (bytes + 255) & ~(size_t)255; return o;
  };
  const size_t ROWF = (size_t)BN_TOT * EMB * sizeof(float);  // 2 MB
  float* f_h   = (float*)(ws + alloc(ROWF));
  float* f_agg = (float*)(ws + alloc(ROWF));
  float* f_xp  = (float*)(ws + alloc(ROWF));
  float* f_x1  = (float*)(ws + alloc(ROWF));
  float* f_q   = (float*)(ws + alloc(ROWF));
  float* f_k   = (float*)(ws + alloc(ROWF));
  float* f_v   = (float*)(ws + alloc(ROWF));
  float* f_o   = (float*)(ws + alloc(ROWF));
  float* f_att = (float*)(ws + alloc(ROWF));
  float* f_ff1 = (float*)(ws + alloc((size_t)BN_TOT * 256 * sizeof(float)));
  float* f_ff2 = (float*)(ws + alloc(ROWF));
  float* f_ea2 = (float*)(ws + alloc((size_t)NEDGE * EIN2 * sizeof(float)));
  float* f_db  = (float*)(ws + alloc((size_t)NGRAPH * NNODE * NNODE * sizeof(float)));
  float* f_sc  = (float*)(ws + alloc((size_t)NGRAPH * NHEAD * NNODE * NNODE * sizeof(float)));
  _Float16* h_proj = (_Float16*)(ws + alloc((size_t)128 * 64 * 2));
  _Float16* h_B    = (_Float16*)(ws + alloc((size_t)128 * KEDGE * 2));
  _Float16* h_wq   = (_Float16*)(ws + alloc((size_t)NLAYER * 128 * 128 * 2));
  _Float16* h_wk   = (_Float16*)(ws + alloc((size_t)NLAYER * 128 * 128 * 2));
  _Float16* h_wv   = (_Float16*)(ws + alloc((size_t)NLAYER * 128 * 128 * 2));
  _Float16* h_wo   = (_Float16*)(ws + alloc((size_t)NLAYER * 128 * 128 * 2));
  _Float16* h_w1   = (_Float16*)(ws + alloc((size_t)NLAYER * 256 * 128 * 2));
  _Float16* h_w2   = (_Float16*)(ws + alloc((size_t)NLAYER * 128 * 256 * 2));

  dim3 b16(16, 16);
  // --- weight prep (f32 -> transposed f16) ---
  transpose_conv_kernel<<<dim3(8, 4), b16, 0, stream>>>(Wproj, h_proj, 64, 128);
  build_edgeB_kernel<<<(128 * KEDGE) / 256, 256, 0, stream>>>(Wlin, blin, h_B);
  for (int l = 0; l < NLAYER; ++l) {
    transpose_conv_kernel<<<dim3(8, 8), b16, 0, stream>>>(Wq + l * 16384, h_wq + l * 16384, 128, 128);
    transpose_conv_kernel<<<dim3(8, 8), b16, 0, stream>>>(Wk + l * 16384, h_wk + l * 16384, 128, 128);
    transpose_conv_kernel<<<dim3(8, 8), b16, 0, stream>>>(Wv + l * 16384, h_wv + l * 16384, 128, 128);
    transpose_conv_kernel<<<dim3(8, 8), b16, 0, stream>>>(Wo + l * 16384, h_wo + l * 16384, 128, 128);
    transpose_conv_kernel<<<dim3(16, 8), b16, 0, stream>>>(W1 + l * 32768, h_w1 + l * 32768, 128, 256);
    transpose_conv_kernel<<<dim3(8, 16), b16, 0, stream>>>(W2 + l * 32768, h_w2 + l * 32768, 256, 128);
  }
  // --- front end ---
  rbf_kernel<<<NEDGE / 256, 256, 0, stream>>>(ea, f_ea2);
  zero_kernel<<<(BN_TOT * EMB) / 256, 256, 0, stream>>>(f_agg, BN_TOT * EMB);
  gemm_kernel<false><<<dim3(2, 64), 128, 0, stream>>>(x, h_proj, bproj, f_h, 64, 128);
  edge_msg_kernel<<<NEDGE / 64, 128, 0, stream>>>(f_ea2, f_h, ei, h_B, f_agg);
  add_ln_kernel<<<BN_TOT / 8, 256, 0, stream>>>(f_h, f_agg, g_en, b_en, f_xp);
  pairdist_kernel<<<(NGRAPH * NNODE * NNODE) / 256, 256, 0, stream>>>(pos, f_db);
  // --- transformer layers ---
  for (int l = 0; l < NLAYER; ++l) {
    gemm_kernel<false><<<dim3(2, 64), 128, 0, stream>>>(f_xp, h_wq + l * 16384, bq + l * 128, f_q, 128, 128);
    gemm_kernel<false><<<dim3(2, 64), 128, 0, stream>>>(f_xp, h_wk + l * 16384, bk + l * 128, f_k, 128, 128);
    gemm_kernel<false><<<dim3(2, 64), 128, 0, stream>>>(f_xp, h_wv + l * 16384, bv + l * 128, f_v, 128, 128);
    attn_kernel<<<NGRAPH * NHEAD, 128, 0, stream>>>(f_q, f_k, f_v, f_db, f_sc, f_o);
    gemm_kernel<false><<<dim3(2, 64), 128, 0, stream>>>(f_o, h_wo + l * 16384, bo + l * 128, f_att, 128, 128);
    add_ln_kernel<<<BN_TOT / 8, 256, 0, stream>>>(f_xp, f_att, g1 + l * 128, b1 + l * 128, f_x1);
    gemm_kernel<true><<<dim3(4, 64), 128, 0, stream>>>(f_x1, h_w1 + l * 32768, bf1 + l * 256, f_ff1, 128, 256);
    gemm_kernel<false><<<dim3(2, 64), 128, 0, stream>>>(f_ff1, h_w2 + l * 32768, bf2 + l * 128, f_ff2, 256, 128);
    add_ln_kernel<<<BN_TOT / 8, 256, 0, stream>>>(f_x1, f_ff2, g2 + l * 128, b2 + l * 128, f_xp);
  }
  pool_kernel<<<NGRAPH, 128, 0, stream>>>(f_xp, (float*)d_out);
}
